// SelfAttention_70342974374089
// MI455X (gfx1250) — matrix operand order repaired and checked by reading, not compiled, not run on hardware
//
#include <hip/hip_runtime.h>
#include <hip/hip_bf16.h>

typedef __attribute__((ext_vector_type(16))) _Float16 v16h;
typedef __attribute__((ext_vector_type(8)))  _Float16 v8h;
typedef __attribute__((ext_vector_type(8)))  float    v8f;
typedef __attribute__((ext_vector_type(4)))  unsigned int v4u;
typedef __attribute__((ext_vector_type(8)))  int          v8i;
typedef __attribute__((ext_vector_type(4)))  int          v4i;

#define DIMC   768
#define HEADS  12
#define HD     64
#define BATCH  4
#define SEQ    2048
#define MTOT   (BATCH * SEQ)   /* 8192 */
#define QKVN   (3 * DIMC)      /* 2304 */

// ---------------------------------------------------------------- WMMA helpers

__device__ __forceinline__ v8f wmma_f16(v16h a, v16h b, v8f c) {
  return __builtin_amdgcn_wmma_f32_16x16x32_f16(
      /*neg_a=*/false, a, /*neg_b=*/false, b,
      /*c_mod=*/(short)0, c, /*reuse_a=*/false, /*reuse_b=*/false);
}

// A operand (16x32 f16): lane%16 = row M; halves 0-7 -> K kb..kb+7,
// halves 8-15 -> K kb+16..kb+23, kb = (lane<16)?0:8. Source row-major, K contig.
__device__ __forceinline__ v16h load_frag_a(const _Float16* base, int stride) {
  const int lane = threadIdx.x & 31;
  const _Float16* p = base + (lane & 15) * stride + ((lane >> 4) << 3);
  v8h lo = *(const v8h*)p;
  v8h hi = *(const v8h*)(p + 16);
  return __builtin_shufflevector(lo, hi, 0,1,2,3,4,5,6,7,8,9,10,11,12,13,14,15);
}

// B operand (32x16 f16) from N-major storage: lane%16 = column N;
// halves 0-7 -> K kb..kb+7, halves 8-15 -> K kb+16..kb+23, kb = (lane<16)?0:8.
__device__ __forceinline__ v16h load_frag_b(const _Float16* base, int stride) {
  const int lane = threadIdx.x & 31;
  const _Float16* p = base + (lane & 15) * stride + ((lane >> 4) << 3);
  v8h lo = *(const v8h*)p;
  v8h hi = *(const v8h*)(p + 16);
  return __builtin_shufflevector(lo, hi, 0,1,2,3,4,5,6,7,8,9,10,11,12,13,14,15);
}

__device__ __forceinline__ float half16_max(float v) {
  v = fmaxf(v, __shfl_xor(v, 1));
  v = fmaxf(v, __shfl_xor(v, 2));
  v = fmaxf(v, __shfl_xor(v, 4));
  v = fmaxf(v, __shfl_xor(v, 8));
  return v;
}
__device__ __forceinline__ float half16_sum(float v) {
  v += __shfl_xor(v, 1);
  v += __shfl_xor(v, 2);
  v += __shfl_xor(v, 4);
  v += __shfl_xor(v, 8);
  return v;
}

// ---------------------------------------------------------------- TDM helper
// 2-D tile load via Tensor Data Mover (D# per CDNA5 ISA ch.8).
// dims/strides in elements (data_size = 2 bytes), addresses in bytes.
__device__ __forceinline__ void tdm_load_2d(unsigned lds_off, const _Float16* gptr,
                                            unsigned tensor_d0, unsigned tensor_d1,
                                            unsigned stride0,
                                            unsigned tile_d0, unsigned tile_d1) {
  const unsigned long long ga = (unsigned long long)(uintptr_t)gptr;
  v4u g0;
  g0[0] = 1u;                                            // count=1 (valid), user mode
  g0[1] = lds_off;                                       // lds_addr (bytes)
  g0[2] = (unsigned)(ga & 0xffffffffu);                  // global_addr[31:0]
  g0[3] = (unsigned)((ga >> 32) & 0x1ffffffu) | (2u << 30); // addr[56:32] | type=2
  v8i g1;
  g1[0] = (int)(1u << 16);                               // data_size=1 -> 2 bytes
  g1[1] = (int)((tensor_d0 & 0xffffu) << 16);            // tensor_dim0[15:0]
  g1[2] = (int)(((tensor_d0 >> 16) & 0xffffu) | ((tensor_d1 & 0xffffu) << 16));
  g1[3] = (int)(((tensor_d1 >> 16) & 0xffffu) | ((tile_d0 & 0xffffu) << 16));
  g1[4] = (int)(tile_d1 & 0xffffu);                      // tile_dim1; tile_dim2=0
  g1[5] = (int)stride0;                                  // tensor_dim0_stride[31:0]
  g1[6] = 0;
  g1[7] = 0;
  const v4i z4 = {0, 0, 0, 0};
  const v8i z8 = {0, 0, 0, 0, 0, 0, 0, 0};
  __builtin_amdgcn_tensor_load_to_lds(g0, g1, z4, z4, z8, 0);
}

// ---------------------------------------------------------------- converters

__global__ __launch_bounds__(256) void cvt_f32_to_f16(const float* __restrict__ in,
                                                      _Float16* __restrict__ out, int n) {
  int i = blockIdx.x * 256 + threadIdx.x;
  if (i < n) out[i] = (_Float16)in[i];
}

// in: [K][N] f32 row-major  ->  out: [N][K] f16 (N-major)
__global__ __launch_bounds__(256) void cvt_transpose_f16(const float* __restrict__ in,
                                                         _Float16* __restrict__ out,
                                                         int K, int N) {
  int i = blockIdx.x * 256 + threadIdx.x;
  if (i < K * N) {
    int k = i / N, n = i - k * N;
    out[(size_t)n * K + k] = (_Float16)in[i];
  }
}

// ---------------------------------------------------------------- QKV GEMM
// C[8192][2304] = xh[8192][768] @ w_qkv + b_qkv ; scatter into Q,K (row) / V (transposed)
// Tiles staged by TDM, double-buffered.

__global__ __launch_bounds__(256) void qkv_gemm(const _Float16* __restrict__ Ah,
                                                const _Float16* __restrict__ BT,
                                                const float* __restrict__ bias,
                                                _Float16* __restrict__ Qb,
                                                _Float16* __restrict__ Kb,
                                                _Float16* __restrict__ Vt) {
  __shared__ _Float16 sA[2][128 * 32];
  __shared__ _Float16 sB[2][128 * 32];
  const int tid = threadIdx.x;
  const int wave = tid >> 5, lane = tid & 31;
  const int wm = wave >> 1, wn = wave & 1;        // 4x2 wave grid: 32x64 per wave
  const int row0 = blockIdx.x * 128, col0 = blockIdx.y * 128;

  v8f acc[2][4];
#pragma unroll
  for (int mt = 0; mt < 2; ++mt)
#pragma unroll
    for (int nt = 0; nt < 4; ++nt) acc[mt][nt] = v8f{};

  const unsigned ldsA0 = (unsigned)(uintptr_t)&sA[0][0];
  const unsigned ldsA1 = (unsigned)(uintptr_t)&sA[1][0];
  const unsigned ldsB0 = (unsigned)(uintptr_t)&sB[0][0];
  const unsigned ldsB1 = (unsigned)(uintptr_t)&sB[1][0];

  if (wave == 0) {
    tdm_load_2d(ldsA0, Ah + (size_t)row0 * DIMC, DIMC, MTOT, DIMC, 32, 128);
    tdm_load_2d(ldsB0, BT + (size_t)col0 * DIMC, DIMC, QKVN, DIMC, 32, 128);
    __builtin_amdgcn_s_wait_tensorcnt(0);
  }
  __syncthreads();

  for (int k0 = 0; k0 < DIMC; k0 += 32) {
    const int cur = (k0 >> 5) & 1, nxt = cur ^ 1;
    const bool more = (k0 + 32) < DIMC;
    if (wave == 0 && more) {
      tdm_load_2d(nxt ? ldsA1 : ldsA0, Ah + (size_t)row0 * DIMC + k0 + 32,
                  DIMC, MTOT, DIMC, 32, 128);
      tdm_load_2d(nxt ? ldsB1 : ldsB0, BT + (size_t)col0 * DIMC + k0 + 32,
                  DIMC, QKVN, DIMC, 32, 128);
    }
    v16h af[2], bf[4];
#pragma unroll
    for (int mt = 0; mt < 2; ++mt) af[mt] = load_frag_a(sA[cur] + (wm * 32 + mt * 16) * 32, 32);
#pragma unroll
    for (int nt = 0; nt < 4; ++nt) bf[nt] = load_frag_b(sB[cur] + (wn * 64 + nt * 16) * 32, 32);
#pragma unroll
    for (int mt = 0; mt < 2; ++mt)
#pragma unroll
      for (int nt = 0; nt < 4; ++nt) acc[mt][nt] = wmma_f16(af[mt], bf[nt], acc[mt][nt]);
    __syncthreads();
    if (wave == 0 && more) __builtin_amdgcn_s_wait_tensorcnt(0);
    __syncthreads();
  }

  // Epilogue: bias + scatter per head. j = global col in [0,2304)
  const int rsub = (lane >> 4) << 3;  // 0 or 8
  const int cl = lane & 15;
#pragma unroll
  for (int mt = 0; mt < 2; ++mt) {
#pragma unroll
    for (int nt = 0; nt < 4; ++nt) {
      const int j = col0 + wn * 64 + nt * 16 + cl;
      const int s = (j >= 2 * DIMC) ? 2 : ((j >= DIMC) ? 1 : 0);
      const int rem = j - s * DIMC;
      const int h = rem >> 6, d = rem & 63;
      const float bj = bias[j];
#pragma unroll
      for (int r = 0; r < 8; ++r) {
        const int row = row0 + wm * 32 + mt * 16 + rsub + r;
        const int b = row >> 11, n = row & 2047;
        const size_t bh = (size_t)(b * HEADS + h);
        const _Float16 hv = (_Float16)(acc[mt][nt][r] + bj);
        if (s == 0)      Qb[(bh * SEQ + n) * HD + d] = hv;
        else if (s == 1) Kb[(bh * SEQ + n) * HD + d] = hv;
        else             Vt[(bh * HD + d) * SEQ + n] = hv;   // V stored transposed
      }
    }
  }
}

// ---------------------------------------------------------------- flash attention
// 8 waves/block share one (b,h); K and V^T tiles TDM-staged into double-buffered
// LDS. Each wave owns 16 Q rows. V pre-transposed [bh][d][seq].

__global__ __launch_bounds__(256) void attn_flash(const _Float16* __restrict__ Qb,
                                                  const _Float16* __restrict__ Kb,
                                                  const _Float16* __restrict__ Vt,
                                                  _Float16* __restrict__ Oh) {
  __shared__ _Float16 sK[2][32 * 64];   // [tile row n][d]
  __shared__ _Float16 sV[2][64 * 32];   // [d][tile col k]
  __shared__ _Float16 sP[8][16 * 32];   // per-wave P staging (C-layout -> A-layout)
  const int wave = threadIdx.x >> 5, lane = threadIdx.x & 31;
  const int bh = blockIdx.x >> 4;                    // 0..47, shared by block
  const int rb = ((blockIdx.x & 15) << 3) + wave;    // 0..127 row-block
  const int b = bh / HEADS, h = bh - b * HEADS;
  const _Float16* Qp = Qb + (size_t)bh * SEQ * HD;
  const _Float16* Kp = Kb + (size_t)bh * SEQ * HD;
  const _Float16* Vp = Vt + (size_t)bh * HD * SEQ;
  _Float16* sPw = sP[wave];
  const int n0 = rb * 16;
  const int rsub = (lane >> 4) << 3, cl = lane & 15;

  const unsigned ldsK0 = (unsigned)(uintptr_t)&sK[0][0];
  const unsigned ldsK1 = (unsigned)(uintptr_t)&sK[1][0];
  const unsigned ldsV0 = (unsigned)(uintptr_t)&sV[0][0];
  const unsigned ldsV1 = (unsigned)(uintptr_t)&sV[1][0];

  const v16h qf0 = load_frag_a(Qp + (size_t)n0 * HD, HD);
  const v16h qf1 = load_frag_a(Qp + (size_t)n0 * HD + 32, HD);

  v8f o0 = v8f{}, o1 = v8f{}, o2 = v8f{}, o3 = v8f{};
  float m[8], l[8];
#pragma unroll
  for (int r = 0; r < 8; ++r) { m[r] = -3.0e38f; l[r] = 0.0f; }

  if (wave == 0) {
    tdm_load_2d(ldsK0, Kp, HD, SEQ, HD, 64, 32);         // K rows 0..31
    tdm_load_2d(ldsV0, Vp, SEQ, HD, SEQ, 32, 64);        // V^T cols 0..31
    __builtin_amdgcn_s_wait_tensorcnt(0);
  }
  __syncthreads();

  for (int j = 0; j < SEQ; j += 32) {
    const int cur = (j >> 5) & 1, nxt = cur ^ 1;
    const bool more = (j + 32) < SEQ;
    if (wave == 0 && more) {
      tdm_load_2d(nxt ? ldsK1 : ldsK0, Kp + (size_t)(j + 32) * HD, HD, SEQ, HD, 64, 32);
      tdm_load_2d(nxt ? ldsV1 : ldsV0, Vp + (size_t)(j + 32), SEQ, HD, SEQ, 32, 64);
    }
    const _Float16* sKc = sK[cur];
    const _Float16* sVc = sV[cur];

    // S = Q @ K^T (two 16-col tiles, K-dim = 64 = two WMMA k-steps)
    v8f s0 = v8f{}, s1 = v8f{};
    {
      v16h kf = load_frag_b(sKc, 64);
      s0 = wmma_f16(qf0, kf, s0);
      kf = load_frag_b(sKc + 32, 64);
      s0 = wmma_f16(qf1, kf, s0);
      kf = load_frag_b(sKc + 16 * 64, 64);
      s1 = wmma_f16(qf0, kf, s1);
      kf = load_frag_b(sKc + 16 * 64 + 32, 64);
      s1 = wmma_f16(qf1, kf, s1);
    }
    // online softmax update (8 rows per lane-half, columns spread over 16 lanes)
#pragma unroll
    for (int r = 0; r < 8; ++r) {
      const float a0 = s0[r] * 0.125f;    // 1/sqrt(64)
      const float a1 = s1[r] * 0.125f;
      const float mx = half16_max(fmaxf(a0, a1));
      const float newm = fmaxf(m[r], mx);
      const float corr = __expf(m[r] - newm);
      const float p0 = __expf(a0 - newm);
      const float p1 = __expf(a1 - newm);
      const float ps = half16_sum(p0 + p1);
      l[r] = l[r] * corr + ps;
      m[r] = newm;
      o0[r] *= corr; o1[r] *= corr; o2[r] *= corr; o3[r] *= corr;
      const int rr = rsub + r;
      sPw[rr * 32 + cl]      = (_Float16)p0;
      sPw[rr * 32 + 16 + cl] = (_Float16)p1;
    }
    asm volatile("s_wait_dscnt 0" ::: "memory");  // P visible across lanes (same-wave LDS order)
    const v16h pf = load_frag_a(sPw, 32);
    o0 = wmma_f16(pf, load_frag_b(sVc + 0 * 16 * 32, 32), o0);
    o1 = wmma_f16(pf, load_frag_b(sVc + 1 * 16 * 32, 32), o1);
    o2 = wmma_f16(pf, load_frag_b(sVc + 2 * 16 * 32, 32), o2);
    o3 = wmma_f16(pf, load_frag_b(sVc + 3 * 16 * 32, 32), o3);

    __syncthreads();                                   // all waves done with buf cur
    if (wave == 0 && more) __builtin_amdgcn_s_wait_tensorcnt(0);
    __syncthreads();                                   // buf nxt ready for all
  }

  // normalize + store f16 into attn_out row-major [B*N][C]
  const size_t rowg = (size_t)b * SEQ + n0;
#pragma unroll
  for (int r = 0; r < 8; ++r) {
    const float inv = 1.0f / l[r];
    const size_t base = (rowg + rsub + r) * DIMC + h * HD + cl;
    Oh[base + 0]  = (_Float16)(o0[r] * inv);
    Oh[base + 16] = (_Float16)(o1[r] * inv);
    Oh[base + 32] = (_Float16)(o2[r] * inv);
    Oh[base + 48] = (_Float16)(o3[r] * inv);
  }
}

// ---------------------------------------------------------------- output projection
// out[8192][768] f32 = attn_h @ w_proj + b_proj ; TDM-staged tiles.

__global__ __launch_bounds__(256) void proj_gemm(const _Float16* __restrict__ Ah,
                                                 const _Float16* __restrict__ BT,
                                                 const float* __restrict__ bias,
                                                 float* __restrict__ out) {
  __shared__ _Float16 sA[2][128 * 32];
  __shared__ _Float16 sB[2][128 * 32];
  const int tid = threadIdx.x;
  const int wave = tid >> 5, lane = tid & 31;
  const int wm = wave >> 1, wn = wave & 1;
  const int row0 = blockIdx.x * 128, col0 = blockIdx.y * 128;

  v8f acc[2][4];
#pragma unroll
  for (int mt = 0; mt < 2; ++mt)
#pragma unroll
    for (int nt = 0; nt < 4; ++nt) acc[mt][nt] = v8f{};

  const unsigned ldsA0 = (unsigned)(uintptr_t)&sA[0][0];
  const unsigned ldsA1 = (unsigned)(uintptr_t)&sA[1][0];
  const unsigned ldsB0 = (unsigned)(uintptr_t)&sB[0][0];
  const unsigned ldsB1 = (unsigned)(uintptr_t)&sB[1][0];

  if (wave == 0) {
    tdm_load_2d(ldsA0, Ah + (size_t)row0 * DIMC, DIMC, MTOT, DIMC, 32, 128);
    tdm_load_2d(ldsB0, BT + (size_t)col0 * DIMC, DIMC, DIMC, DIMC, 32, 128);
    __builtin_amdgcn_s_wait_tensorcnt(0);
  }
  __syncthreads();

  for (int k0 = 0; k0 < DIMC; k0 += 32) {
    const int cur = (k0 >> 5) & 1, nxt = cur ^ 1;
    const bool more = (k0 + 32) < DIMC;
    if (wave == 0 && more) {
      tdm_load_2d(nxt ? ldsA1 : ldsA0, Ah + (size_t)row0 * DIMC + k0 + 32,
                  DIMC, MTOT, DIMC, 32, 128);
      tdm_load_2d(nxt ? ldsB1 : ldsB0, BT + (size_t)col0 * DIMC + k0 + 32,
                  DIMC, DIMC, DIMC, 32, 128);
    }
    v16h af[2], bf[4];
#pragma unroll
    for (int mt = 0; mt < 2; ++mt) af[mt] = load_frag_a(sA[cur] + (wm * 32 + mt * 16) * 32, 32);
#pragma unroll
    for (int nt = 0; nt < 4; ++nt) bf[nt] = load_frag_b(sB[cur] + (wn * 64 + nt * 16) * 32, 32);
#pragma unroll
    for (int mt = 0; mt < 2; ++mt)
#pragma unroll
      for (int nt = 0; nt < 4; ++nt) acc[mt][nt] = wmma_f16(af[mt], bf[nt], acc[mt][nt]);
    __syncthreads();
    if (wave == 0 && more) __builtin_amdgcn_s_wait_tensorcnt(0);
    __syncthreads();
  }

  const int rsub = (lane >> 4) << 3;
  const int cl = lane & 15;
#pragma unroll
  for (int mt = 0; mt < 2; ++mt) {
#pragma unroll
    for (int nt = 0; nt < 4; ++nt) {
      const int j = col0 + wn * 64 + nt * 16 + cl;
      const float bj = bias[j];
#pragma unroll
      for (int r = 0; r < 8; ++r) {
        const int row = row0 + wm * 32 + mt * 16 + rsub + r;
        out[(size_t)row * DIMC + j] = acc[mt][nt][r] + bj;
      }
    }
  }
}

// ---------------------------------------------------------------- launch

extern "C" void kernel_launch(void* const* d_in, const int* in_sizes, int n_in,
                              void* d_out, int out_size, void* d_ws, size_t ws_size,
                              hipStream_t stream) {
  const float* x      = (const float*)d_in[0];
  const float* w_qkv  = (const float*)d_in[1];
  const float* b_qkv  = (const float*)d_in[2];
  const float* w_proj = (const float*)d_in[3];
  const float* b_proj = (const float*)d_in[4];
  float* out = (float*)d_out;

  const size_t NXC = (size_t)MTOT * DIMC;       // 6291456
  char* ws = (char*)d_ws;
  _Float16* xh     = (_Float16*)ws;  ws += NXC * 2;
  _Float16* wqkvT  = (_Float16*)ws;  ws += (size_t)DIMC * QKVN * 2;
  _Float16* wprojT = (_Float16*)ws;  ws += (size_t)DIMC * DIMC * 2;
  _Float16* Qb     = (_Float16*)ws;  ws += NXC * 2;
  _Float16* Kb     = (_Float16*)ws;  ws += NXC * 2;
  _Float16* Vt     = (_Float16*)ws;  ws += NXC * 2;
  _Float16* attn_h = (_Float16*)ws;  ws += NXC * 2;

  // 1) convert / transpose inputs to f16
  cvt_f32_to_f16<<<(int)((NXC + 255) / 256), 256, 0, stream>>>(x, xh, (int)NXC);
  cvt_transpose_f16<<<(DIMC * QKVN + 255) / 256, 256, 0, stream>>>(w_qkv, wqkvT, DIMC, QKVN);
  cvt_transpose_f16<<<(DIMC * DIMC + 255) / 256, 256, 0, stream>>>(w_proj, wprojT, DIMC, DIMC);

  // 2) QKV projection: [8192,768] @ [768,2304] -> Q,K,Vt
  qkv_gemm<<<dim3(MTOT / 128, QKVN / 128), 256, 0, stream>>>(xh, wqkvT, b_qkv, Qb, Kb, Vt);

  // 3) flash attention: 48 heads x 16 blocks, 8 waves/block share (b,h)
  attn_flash<<<BATCH * HEADS * 16, 256, 0, stream>>>(Qb, Kb, Vt, attn_h);

  // 4) output projection: [8192,768] @ [768,768] + bias -> f32 out
  proj_gemm<<<dim3(MTOT / 128, DIMC / 128), 256, 0, stream>>>(attn_h, wprojT, b_proj, out);
}
